// Cluster_Level_GCN_63745904607645
// MI455X (gfx1250) — matrix operand order repaired and checked by reading, not compiled, not run on hardware
//
#include <hip/hip_runtime.h>
#include <hip/hip_bf16.h>
#include <math.h>

// ---------------- problem constants ----------------
constexpr int B    = 64;
constexpr int K    = 64;
constexpr int N    = 32768;
constexpr int D    = 2048;
constexpr int NC   = 2048;
constexpr int NHID = 512;
constexpr int TOPK = 5;
constexpr int BK   = B * K;          // 4096 rows
constexpr int D2   = 2 * D;          // 4096 concat width

// ---------------- WMMA vector types ----------------
typedef float  v2f   __attribute__((ext_vector_type(2)));
typedef float  v8f   __attribute__((ext_vector_type(8)));
typedef __bf16 v16bf __attribute__((ext_vector_type(16)));

// ---------------- CDNA5 async copy helpers ----------------
// GLOBAL_LOAD_ASYNC_TO_LDS_B128: per-lane 16B global -> LDS, tracked by ASYNCcnt.
__device__ __forceinline__ void async_copy_b128(unsigned lds_off, const void* gptr) {
    unsigned long long ga = (unsigned long long)(uintptr_t)gptr;
    asm volatile("global_load_async_to_lds_b128 %0, %1, off"
                 :: "v"(lds_off), "v"(ga) : "memory");
}
__device__ __forceinline__ void wait_async_le(int n) {
    if (n == 0) asm volatile("s_wait_asynccnt 0x0" ::: "memory");
    else        asm volatile("s_wait_asynccnt 0x3" ::: "memory");
}

// =====================================================================
// 0) zero workspace region (cluster sums + counts) -- ws is poisoned
// =====================================================================
__global__ void zero_kernel(float* p, size_t n) {
    size_t i = (size_t)blockIdx.x * blockDim.x + threadIdx.x;
    size_t stride = (size_t)gridDim.x * blockDim.x;
    for (; i < n; i += stride) p[i] = 0.0f;
}

// =====================================================================
// 1) cluster segment-sum via float atomics (global_atomic_add_f32)
// =====================================================================
__global__ void cluster_accum_kernel(const float* __restrict__ features,
                                     const int* __restrict__ labels,
                                     float* __restrict__ clu_sum,
                                     float* __restrict__ counts) {
    int n   = blockIdx.x;
    int lab = labels[n];
    const float* src = features + (size_t)n * D;
    float*       dst = clu_sum + (size_t)lab * D;
    for (int d = threadIdx.x; d < D; d += blockDim.x)
        unsafeAtomicAdd(&dst[d], src[d]);
    if (threadIdx.x == 0) unsafeAtomicAdd(&counts[lab], 1.0f);
}

__global__ void cluster_div_kernel(float* __restrict__ clu_mean,
                                   const float* __restrict__ counts) {
    size_t idx = (size_t)blockIdx.x * blockDim.x + threadIdx.x;
    if (idx >= (size_t)NC * D) return;
    int c = (int)(idx / D);
    clu_mean[idx] /= fmaxf(counts[c], 1.0f);
}

// =====================================================================
// 2) gather per-graph node features; row 0 = query's raw feature
// =====================================================================
__global__ void gather_kernel(const float* __restrict__ features,
                              const float* __restrict__ clu_mean,
                              const int* __restrict__ labels,
                              const int* __restrict__ indexes,
                              const int* __restrict__ nbr,
                              float* __restrict__ cf,
                              int* __restrict__ clu_lab) {
    int k = blockIdx.x, b = blockIdx.y;
    int nb  = nbr[b * K + k];
    int lab = labels[nb];
    if (threadIdx.x == 0) clu_lab[b * K + k] = lab;
    const float* src = (k == 0) ? (features + (size_t)indexes[b] * D)
                                : (clu_mean + (size_t)lab * D);
    float* dst = cf + (size_t)(b * K + k) * D;
    for (int d = threadIdx.x; d < D; d += blockDim.x) dst[d] = src[d];
}

// =====================================================================
// 3) affinity A = cf * cf^T / 5  (f32 WMMA 16x16x4)
//    One workgroup per graph: stage 64x64-float K-chunks of cf in LDS via
//    async copies (double-buffered); 8 waves x 2 tiles = all 16 tiles.
// =====================================================================
constexpr int AF_PAD = 68;            // 64 + 4 floats pad (bank spread, 16B mult)

__global__ void affinity_kernel(const float* __restrict__ cf,
                                float* __restrict__ A) {
    __shared__ __align__(16) float cfs[2][K * AF_PAD];   // 2 x 17408 B
    int b    = blockIdx.x;
    int tid  = threadIdx.x;           // 256
    int wave = tid >> 5, lane = tid & 31;
    int half = lane >> 4, l = lane & 15;
    int t0 = wave * 2;                // this wave's tiles: t0, t0+1 (share tm)
    int tm = t0 >> 2;
    const float* base = cf + (size_t)b * K * D;

    auto issue = [&](int buf, int k0) {
        // 64 rows x 16 chunks (16B) = 1024 chunks; 4 per thread
#pragma unroll
        for (int q = 0; q < 4; q++) {
            int c   = tid + q * 256;
            int row = c >> 4, cc = c & 15;
            async_copy_b128((unsigned)(uintptr_t)&cfs[buf][row * AF_PAD + cc * 4],
                            base + (size_t)row * D + k0 + cc * 4);
        }
    };

    v8f acc[2] = {};
    issue(0, 0);
    constexpr int NSTEP = D / 64;     // 32
    for (int s = 0; s < NSTEP; s++) {
        int buf = s & 1;
        if (s + 1 < NSTEP) { issue(buf ^ 1, (s + 1) * 64); wait_async_le(4); }
        else wait_async_le(0);
        __syncthreads();
#pragma unroll
        for (int k0 = 0; k0 < 64; k0 += 4) {
            int kk = k0 + 2 * half;
            v2f a;
            a.x = cfs[buf][(tm * 16 + l) * AF_PAD + kk];
            a.y = cfs[buf][(tm * 16 + l) * AF_PAD + kk + 1];
#pragma unroll
            for (int jn = 0; jn < 2; jn++) {
                int tn = (t0 + jn) & 3;
                v2f bb;
                bb.x = cfs[buf][(tn * 16 + l) * AF_PAD + kk];
                bb.y = cfs[buf][(tn * 16 + l) * AF_PAD + kk + 1];
                acc[jn] = __builtin_amdgcn_wmma_f32_16x16x4_f32(
                    false, a, false, bb, (short)0, acc[jn], false, false);
            }
        }
        __syncthreads();
    }
#pragma unroll
    for (int jn = 0; jn < 2; jn++) {
        int tn = (t0 + jn) & 3;
#pragma unroll
        for (int r = 0; r < 8; r++) {
            int M = tm * 16 + 8 * half + r;
            A[(size_t)b * K * K + M * K + tn * 16 + l] = acc[jn][r] * 0.2f;
        }
    }
}

// =====================================================================
// 4) top-5 symmetric mask + first-occurrence keep; in-place on A
// =====================================================================
__global__ void mask_kernel(float* __restrict__ A,
                            const int* __restrict__ clu_lab) {
    __shared__ float         As[K * K];
    __shared__ int           lab[K];
    __shared__ unsigned char m[K * K];
    __shared__ float         keep[K];
    int b = blockIdx.x;
    int i = threadIdx.x;              // 64 threads

    lab[i] = clu_lab[b * K + i];
    for (int j = 0; j < K; j++) {
        As[i * K + j] = A[(size_t)b * K * K + i * K + j];
        m[i * K + j]  = 0;
    }
    __syncthreads();

    bool dup = false;
    for (int j = 0; j < i; j++) dup |= (lab[j] == lab[i]);
    keep[i] = dup ? 0.0f : 1.0f;

    for (int t = 0; t < TOPK; t++) {
        float best = -INFINITY; int bi = 0;
        for (int j = 0; j < K; j++) {
            if (m[i * K + j]) continue;
            float v = As[i * K + j];
            if (v > best) { best = v; bi = j; }
        }
        m[i * K + bi] = 1;
    }
    __syncthreads();

    for (int j = 0; j < K; j++) {
        float mm = (m[i * K + j] && m[j * K + i]) ? 1.0f : 0.0f;
        A[(size_t)b * K * K + i * K + j] =
            As[i * K + j] * mm * keep[i] * keep[j];
    }
}

// =====================================================================
// 5) row-normalize + center on row 0; write x (f32) and xa[:,0:D] (bf16)
// =====================================================================
__global__ void normalize_kernel(const float* __restrict__ cf,
                                 float* __restrict__ x,
                                 __bf16* __restrict__ xa) {
    int i = blockIdx.x, b = blockIdx.y;
    int tid = threadIdx.x;            // 256 threads
    const float* row  = cf + (size_t)(b * K + i) * D;
    const float* row0 = cf + (size_t)(b * K) * D;

    float s = 0.f, s0 = 0.f;
    for (int d = tid; d < D; d += 256) {
        float v = row[d];   s  += v * v;
        float v0 = row0[d]; s0 += v0 * v0;
    }
    __shared__ float red[256], red0[256];
    red[tid] = s; red0[tid] = s0;
    __syncthreads();
    for (int off = 128; off > 0; off >>= 1) {
        if (tid < off) { red[tid] += red[tid + off]; red0[tid] += red0[tid + off]; }
        __syncthreads();
    }
    float inv  = rsqrtf(red[0]);
    float inv0 = rsqrtf(red0[0]);

    size_t rowOut = (size_t)(b * K + i);
    for (int d = tid; d < D; d += 256) {
        float v = row[d] * inv - row0[d] * inv0;
        x[rowOut * D + d]   = v;
        xa[rowOut * D2 + d] = (__bf16)v;
    }
}

// =====================================================================
// 6) agg = adj @ x (f32 WMMA 16x16x4) -> xa[:, D:2D] as bf16
//    adj is 16KB/graph (L2-resident); x B-frag loads are lane-coalesced.
// =====================================================================
__global__ void agg_kernel(const float* __restrict__ adj,
                           const float* __restrict__ x,
                           __bf16* __restrict__ xa) {
    int tn = blockIdx.x;              // 0..127 (D/16)
    int tm = blockIdx.y;              // 0..3   (K/16)
    int b  = blockIdx.z;
    int lane = threadIdx.x & 31;
    int half = lane >> 4, l = lane & 15;

    const float* Ab = adj + (size_t)b * K * K;
    const float* Xb = x + (size_t)b * K * D;
    int mRow = tm * 16 + l;
    int nCol = tn * 16 + l;

    v8f acc = {};
    for (int k0 = 0; k0 < K; k0 += 4) {
        int kk = k0 + 2 * half;
        v2f a, bb;
        a.x  = Ab[mRow * K + kk];
        a.y  = Ab[mRow * K + kk + 1];
        bb.x = Xb[(size_t)kk * D + nCol];
        bb.y = Xb[(size_t)(kk + 1) * D + nCol];
        acc = __builtin_amdgcn_wmma_f32_16x16x4_f32(
            false, a, false, bb, (short)0, acc, false, false);
    }
#pragma unroll
    for (int r = 0; r < 8; r++) {
        int M = tm * 16 + 8 * half + r;
        xa[(size_t)(b * K + M) * D2 + D + tn * 16 + l] = (__bf16)acc[r];
    }
}

// =====================================================================
// 7) f32 -> bf16 weight conversion
// =====================================================================
__global__ void f32_to_bf16_kernel(const float* __restrict__ in,
                                   __bf16* __restrict__ out, size_t n) {
    size_t i = (size_t)blockIdx.x * blockDim.x + threadIdx.x;
    size_t stride = (size_t)gridDim.x * blockDim.x;
    for (; i < n; i += stride) out[i] = (__bf16)in[i];
}

// =====================================================================
// 8) dense layers: bf16 WMMA 16x16x32, f32 accumulate
//    Block 256 thr = 8 waves; block tile 128x64; K-step 32; LDS tiles
//    double-buffered via async copies. Wave (4x2 grid) owns 32x32.
//    MODE 0: ReLU -> bf16 out ; MODE 1: PReLU -> f32 out
// =====================================================================
constexpr int GBM = 128, GBN = 64, GBK = 32;
constexpr int ASTRIDE = 40;           // 32 + 8 bf16 pad (80B rows, 16B mult)

template <int MODE>
__global__ void gemm_bf16_kernel(const __bf16* __restrict__ A,
                                 const __bf16* __restrict__ Bm,
                                 const float* __restrict__ bias,
                                 const float* __restrict__ pa,
                                 __bf16* __restrict__ out_bf,
                                 float* __restrict__ out_f,
                                 int Nsz, int Kd) {
    __shared__ __align__(16) __bf16 As[2][GBM * ASTRIDE];  // 2 x 10240 B
    __shared__ __align__(16) __bf16 Bs[2][GBK * GBN];      // 2 x  4096 B
    int tid  = threadIdx.x;
    int wave = tid >> 5, lane = tid & 31;
    int half = lane >> 4, l = lane & 15;
    int waveM = wave >> 1, waveN = wave & 1;
    int mBase = blockIdx.y * GBM;
    int nBase = blockIdx.x * GBN;

    auto issue = [&](int buf, int k0) {
        // A: 128 rows x 4 chunks(16B) = 512; 2 per thread
#pragma unroll
        for (int q = 0; q < 2; q++) {
            int c   = tid * 2 + q;
            int row = c >> 2, cc = c & 3;
            async_copy_b128((unsigned)(uintptr_t)&As[buf][row * ASTRIDE + cc * 8],
                            A + (size_t)(mBase + row) * Kd + k0 + cc * 8);
        }
        // B: 32 rows x 8 chunks(16B) = 256; 1 per thread
        {
            int row = tid >> 3, cc = tid & 7;
            async_copy_b128((unsigned)(uintptr_t)&Bs[buf][row * GBN + cc * 8],
                            Bm + (size_t)(k0 + row) * Nsz + nBase + cc * 8);
        }
    };

    v8f acc[2][2] = {};
    issue(0, 0);
    int nsteps = Kd / GBK;
    for (int s = 0; s < nsteps; s++) {
        int buf = s & 1;
        if (s + 1 < nsteps) { issue(buf ^ 1, (s + 1) * GBK); wait_async_le(3); }
        else wait_async_le(0);
        __syncthreads();
#pragma unroll
        for (int i = 0; i < 2; i++) {
            int mt = waveM * 32 + i * 16;
            v16bf afrag;
            int kA = 8 * half;
#pragma unroll
            for (int j = 0; j < 8; j++) {
                afrag[j]     = As[buf][(mt + l) * ASTRIDE + kA + j];
                afrag[j + 8] = As[buf][(mt + l) * ASTRIDE + 16 + kA + j];
            }
#pragma unroll
            for (int jn = 0; jn < 2; jn++) {
                int nt = waveN * 32 + jn * 16;
                v16bf bfrag;
                int kB = 8 * half;
#pragma unroll
                for (int j = 0; j < 8; j++) {
                    bfrag[j]     = Bs[buf][(kB + j) * GBN + nt + l];
                    bfrag[j + 8] = Bs[buf][(16 + kB + j) * GBN + nt + l];
                }
                acc[i][jn] = __builtin_amdgcn_wmma_f32_16x16x32_bf16(
                    false, afrag, false, bfrag, (short)0, acc[i][jn], false, false);
            }
        }
        __syncthreads();
    }

#pragma unroll
    for (int jn = 0; jn < 2; jn++) {
        int nCol = nBase + waveN * 32 + jn * 16 + l;
        float bv = bias[nCol];
        float av = (MODE == 1) ? pa[nCol] : 0.0f;
#pragma unroll
        for (int i = 0; i < 2; i++) {
#pragma unroll
            for (int r = 0; r < 8; r++) {
                int M = mBase + waveM * 32 + i * 16 + 8 * half + r;
                float v = acc[i][jn][r] + bv;
                if (MODE == 0) {
                    v = v > 0.0f ? v : 0.0f;
                    out_bf[(size_t)M * Nsz + nCol] = (__bf16)v;
                } else {
                    v = v > 0.0f ? v : av * v;
                    out_f[(size_t)M * Nsz + nCol] = v;
                }
            }
        }
    }
}

// =====================================================================
// 9) head: logits = h1 @ w2 + b2, softmax over 2 classes
// =====================================================================
__global__ void head_kernel(const float* __restrict__ h1,
                            const float* __restrict__ w2,
                            const float* __restrict__ b2,
                            float* __restrict__ out) {
    int r = blockIdx.x, t = threadIdx.x;     // 64 threads
    float s0 = 0.f, s1 = 0.f;
    for (int c = t; c < NHID; c += 64) {
        float v = h1[(size_t)r * NHID + c];
        s0 += v * w2[c * 2 + 0];
        s1 += v * w2[c * 2 + 1];
    }
    __shared__ float r0[64], r1[64];
    r0[t] = s0; r1[t] = s1;
    __syncthreads();
    for (int off = 32; off > 0; off >>= 1) {
        if (t < off) { r0[t] += r0[t + off]; r1[t] += r1[t + off]; }
        __syncthreads();
    }
    if (t == 0) {
        float l0 = r0[0] + b2[0], l1 = r1[0] + b2[1];
        float mx = fmaxf(l0, l1);
        float e0 = expf(l0 - mx), e1 = expf(l1 - mx);
        float inv = 1.0f / (e0 + e1);
        out[r * 2 + 0] = e0 * inv;
        out[r * 2 + 1] = e1 * inv;
    }
}

// =====================================================================
// launch
// =====================================================================
extern "C" void kernel_launch(void* const* d_in, const int* in_sizes, int n_in,
                              void* d_out, int out_size, void* d_ws, size_t ws_size,
                              hipStream_t stream) {
    const int*   indexes  = (const int*)  d_in[0];
    const float* features = (const float*)d_in[1];
    const int*   labels   = (const int*)  d_in[2];
    const int*   nbr      = (const int*)  d_in[3];
    const float* conv_w   = (const float*)d_in[4];
    const float* conv_b   = (const float*)d_in[5];
    const float* w1       = (const float*)d_in[6];
    const float* b1       = (const float*)d_in[7];
    const float* prelu_a  = (const float*)d_in[8];
    const float* w2       = (const float*)d_in[9];
    const float* b2       = (const float*)d_in[10];
    float* out = (float*)d_out;

    // ---- workspace carving (256B aligned) ----
    char* ws = (char*)d_ws;
    size_t off = 0;
    auto carve = [&](size_t bytes) {
        void* p = ws + off;
        off += (bytes + 255) & ~(size_t)255;
        return p;
    };
    float*  clu_sum  = (float*) carve((size_t)NC * D * 4);   // -> clu_mean
    float*  counts   = (float*) carve((size_t)NC * 4);
    float*  cf       = (float*) carve((size_t)BK * D * 4);
    float*  x        = (float*) carve((size_t)BK * D * 4);
    int*    clu_lab  = (int*)   carve((size_t)BK * 4);
    float*  adj      = (float*) carve((size_t)B * K * K * 4);
    __bf16* xa       = (__bf16*)carve((size_t)BK * D2 * 2);
    __bf16* convw_bf = (__bf16*)carve((size_t)D2 * NHID * 2);
    __bf16* w1_bf    = (__bf16*)carve((size_t)NHID * NHID * 2);
    __bf16* h_bf     = (__bf16*)carve((size_t)BK * NHID * 2);
    float*  h1       = (float*) carve((size_t)BK * NHID * 4);
    (void)ws_size;

    // 0) zero cluster accumulators
    zero_kernel<<<4096, 256, 0, stream>>>(clu_sum, (size_t)NC * D);
    zero_kernel<<<8, 256, 0, stream>>>(counts, (size_t)NC);

    // 1) cluster means
    cluster_accum_kernel<<<N, 256, 0, stream>>>(features, labels, clu_sum, counts);
    cluster_div_kernel<<<(NC * D) / 256, 256, 0, stream>>>(clu_sum, counts);
    float* clu_mean = clu_sum;

    // weight conversions
    f32_to_bf16_kernel<<<2048, 256, 0, stream>>>(conv_w, convw_bf, (size_t)D2 * NHID);
    f32_to_bf16_kernel<<<512, 256, 0, stream>>>(w1, w1_bf, (size_t)NHID * NHID);

    // 2) gather cf and labels
    gather_kernel<<<dim3(K, B), 256, 0, stream>>>(features, clu_mean, labels,
                                                  indexes, nbr, cf, clu_lab);

    // 3) affinity (f32 WMMA, async LDS staging)
    affinity_kernel<<<B, 256, 0, stream>>>(cf, adj);

    // 4) top-k symmetric mask
    mask_kernel<<<B, K, 0, stream>>>(adj, clu_lab);

    // 5) normalize + center; write x and xa[:, 0:D]
    normalize_kernel<<<dim3(K, B), 256, 0, stream>>>(cf, x, xa);

    // 6) agg = adj @ x (f32 WMMA); write xa[:, D:2D]
    agg_kernel<<<dim3(D / 16, K / 16, B), 32, 0, stream>>>(adj, x, xa);

    // 7) conv layer: h = relu(xa @ conv_w + conv_b)   [4096x4096]x[4096x512]
    gemm_bf16_kernel<0><<<dim3(NHID / GBN, BK / GBM), 256, 0, stream>>>(
        xa, convw_bf, conv_b, nullptr, h_bf, nullptr, NHID, D2);

    // 8) h1 = prelu(h @ w1 + b1)                      [4096x512]x[512x512]
    gemm_bf16_kernel<1><<<dim3(NHID / GBN, BK / GBM), 256, 0, stream>>>(
        h_bf, w1_bf, b1, prelu_a, nullptr, h1, NHID, NHID);

    // 9) logits + softmax
    head_kernel<<<BK, 64, 0, stream>>>(h1, w2, b2, out);
}
